// MultiHeadedAttention_3917010174428
// MI455X (gfx1250) — matrix operand order repaired and checked by reading, not compiled, not run on hardware
//
#include <hip/hip_runtime.h>
#include <hip/hip_bf16.h>

// ---------------------------------------------------------------------------
// MultiHeadedAttention (SuperGlue style) for gfx1250 / MI455X, wave32 WMMA.
//   B=4, D_MODEL=256, N=2048, H=4, head_dim=64.  channel c -> (d=c>>2, h=c&3)
// All GEMMs via v_wmma_f32_16x16x32_f16 (f16 in, f32 acc).
// ---------------------------------------------------------------------------

typedef _Float16 v16h __attribute__((ext_vector_type(16)));
typedef _Float16 v8h  __attribute__((ext_vector_type(8)));
typedef float    v8f  __attribute__((ext_vector_type(8)));

constexpr int NSEQ   = 2048;
constexpr int DMODEL = 256;
constexpr int NHEAD  = 4;
constexpr int HDIM   = 64;
constexpr int BATCH  = 4;

static __device__ __forceinline__ v8f wmma_f16(v16h a, v16h b, v8f c) {
  // (neg_a, A, neg_b, B, c_mod, C, reuse_a, reuse_b)
  return __builtin_amdgcn_wmma_f32_16x16x32_f16(false, a, false, b, (short)0, c,
                                                false, false);
}

static __device__ __forceinline__ v16h cat8(v8h lo, v8h hi) {
  return __builtin_shufflevector(lo, hi, 0, 1, 2, 3, 4, 5, 6, 7, 8, 9, 10, 11,
                                 12, 13, 14, 15);
}

// A-fragment (16x32 f16, M rows = lane&15) from a row-major [row][k] f16 array
// where this lane's row pointer p already includes the (lane>>4)*8 k-offset.
// j=0..7  -> k = kh*8 + j        (contiguous at p)
// j=8..15 -> k = 16 + kh*8 + j-8 (contiguous at p+16)
static __device__ __forceinline__ v16h loadA16(const _Float16* p) {
  v8h lo = *(const v8h*)(p);
  v8h hi = *(const v8h*)(p + 16);
  return cat8(lo, hi);
}

// lane-pair (L <-> L^16) exchange via ds_swizzle (xor_mask=0x10, and=0x1f)
static __device__ __forceinline__ float xor16(float x) {
  int v = __builtin_amdgcn_ds_swizzle(__float_as_int(x), 0x401f);
  return __int_as_float(v);
}

// ---------------------------------------------------------------------------
// Projection: out[o][n] = sum_i W[o][i] * X[b][i][n] + bias[o], cast to f16.
// MODE 0 (Q): scale by 1/sqrt(64) and store d-major  Qd[b][h][d][n]
// MODE 1 (K/V): store n-major  Kt/Vt[b][h][n][d]
// One wave per 16x16 output tile; K=256 fully unrolled (8 WMMA).
// Bias is folded into the C-matrix initialization.
// ---------------------------------------------------------------------------
template <int MODE>
__global__ __launch_bounds__(128) void proj_wmma(const float* __restrict__ X,
                                                 const float* __restrict__ W,
                                                 const float* __restrict__ bias,
                                                 _Float16* __restrict__ out) {
  const int lane = threadIdx.x & 31;
  const int wv   = threadIdx.x >> 5;
  const int m    = lane & 15;
  const int kh   = lane >> 4;
  const int n0   = blockIdx.x * 16;
  const int o0   = (blockIdx.y * 4 + wv) * 16;
  const int b    = blockIdx.z;

  v8f acc;  // C init = bias[o]; D rows o = o0 + kh*8 + r, col n = n0 + m
#pragma unroll
  for (int r = 0; r < 8; ++r) acc[r] = bias[o0 + kh * 8 + r];

  const float* wrow = W + (size_t)(o0 + m) * DMODEL + kh * 8;
  const float* xcol = X + (size_t)b * DMODEL * NSEQ + n0;

#pragma unroll
  for (int k0 = 0; k0 < DMODEL; k0 += 32) {
    v16h a, bf;
#pragma unroll
    for (int v = 0; v < 8; ++v) {  // A: W[o0+m][k], k = grp*16 + kh*8 + 2p + bit
      const int grp = v >> 2, p = v & 3;
      float2 w2 = *(const float2*)(wrow + k0 + grp * 16 + p * 2);
      a[2 * v]     = (_Float16)w2.x;
      a[2 * v + 1] = (_Float16)w2.y;
    }
    const float* xr = xcol + m;  // B col n = n0 + m, k in table order
#pragma unroll
    for (int j = 0; j < 16; ++j) {
      const int kk = k0 + (j >> 3) * 16 + kh * 8 + (j & 7);
      bf[j] = (_Float16)xr[(size_t)kk * NSEQ];
    }
    acc = wmma_f16(a, bf, acc);
  }

#pragma unroll
  for (int r = 0; r < 8; ++r) {
    const int o   = o0 + kh * 8 + r;
    const int h   = o & 3;
    const int d   = o >> 2;
    const size_t bh = (size_t)b * NHEAD + h;
    if (MODE == 0) {  // Q: fold 1/sqrt(head_dim) into q
      out[(bh * HDIM + d) * NSEQ + n0 + m] = (_Float16)(acc[r] * 0.125f);
    } else {          // K/V: n-major rows for A/B fragment loads
      out[(bh * NSEQ + n0 + m) * HDIM + d] = (_Float16)acc[r];
    }
  }
}

// ---------------------------------------------------------------------------
// Flash attention: one wave owns a 16-query strip of one (b,h); streams all
// 2048 keys in chunks of 32.  Per chunk: 4 WMMA for S^T = K^T*Q (two 16-m
// tiles x two K=32 d-slabs), online softmax, 4 WMMA for X^T += P * V^T.
// S^T C/D tiles convert elementwise into the PV A-fragment (no shuffles).
// All K/V fragment loads are hoisted to the top of the iteration so the
// clause issues together and WMMA/softmax drain against partial loadcnt.
// Output: Xf[b][c][n] f16 with c = d*4 + h (merge-kernel layout).
// ---------------------------------------------------------------------------
__global__ __launch_bounds__(256) void attn_wmma(const _Float16* __restrict__ Qd,
                                                 const _Float16* __restrict__ Kt,
                                                 const _Float16* __restrict__ Vt,
                                                 _Float16* __restrict__ Xf) {
  __shared__ float sm[8][16];  // per-wave per-n broadcast slot

  const int lane = threadIdx.x & 31;
  const int wv   = threadIdx.x >> 5;
  const int m16  = lane & 15;
  const int kh   = lane >> 4;
  const int wid  = blockIdx.x * 8 + wv;
  const int ntile = wid & (NSEQ / 16 - 1);
  const int bh    = wid >> 7;  // b*4 + h
  const int n0    = ntile * 16;

  const _Float16* Qbase = Qd + (size_t)bh * HDIM * NSEQ;  // [d][n]
  const _Float16* Kbase = Kt + (size_t)bh * NSEQ * HDIM;  // [m][d]
  const _Float16* Vbase = Vt + (size_t)bh * NSEQ * HDIM;  // [m][d]

  // Q B-fragments for the two d-slabs (col n = n0 + m16, k = d in table order); reused.
  v16h bq0, bq1;
#pragma unroll
  for (int j = 0; j < 16; ++j) {
    const int kd = (j >> 3) * 16 + kh * 8 + (j & 7);
    bq0[j] = Qbase[(size_t)kd * NSEQ + n0 + m16];
    bq1[j] = Qbase[(size_t)(32 + kd) * NSEQ + n0 + m16];
  }

  v8f acc[4] = {};                     // X^T tiles: [n rows][d cols], 4 d-tiles
  float mrow = -INFINITY, lrow = 0.0f; // stats for n = n0 + m16 (lane-pair dup)

  for (int mt = 0; mt < NSEQ; mt += 32) {
    // ---- issue ALL global fragment loads for this chunk up front ----
    const _Float16* ka   = Kbase + (size_t)(mt + m16) * HDIM + kh * 8;
    const _Float16* kb   = ka + 16 * HDIM;
    const _Float16* vrow = Vbase + (size_t)mt * HDIM + m16;
    __builtin_prefetch(ka + 32 * HDIM, 0, 3);              // next K chunk
    __builtin_prefetch(vrow + 32 * HDIM, 0, 3);            // next V chunk
    const v16h a0 = loadA16(ka);
    const v16h a1 = loadA16(kb);
    const v16h a2 = loadA16(ka + 32);
    const v16h a3 = loadA16(kb + 32);
    v16h vb0, vb1, vb2, vb3;
#pragma unroll
    for (int j = 0; j < 16; ++j) {
      const _Float16* vp = vrow + (size_t)((j >> 3) * 16 + kh * 8 + (j & 7)) * HDIM;
      vb0[j] = vp[0];
      vb1[j] = vp[16];
      vb2[j] = vp[32];
      vb3[j] = vp[48];
    }

    // ---- S^T tiles (M = m keys, N = n queries) ----
    v8f s0 = {}, s1 = {};
    s0 = wmma_f16(a0, bq0, s0);
    s1 = wmma_f16(a1, bq0, s1);
    s0 = wmma_f16(a2, bq1, s0);
    s1 = wmma_f16(a3, bq1, s1);

    // ---- online softmax along m (per n column = per lane-pair) ----
    float tmax = -INFINITY;
#pragma unroll
    for (int r = 0; r < 8; ++r) tmax = fmaxf(tmax, fmaxf(s0[r], s1[r]));
    tmax = fmaxf(tmax, xor16(tmax));
    const float mnew  = fmaxf(mrow, tmax);
    const float scale = __expf(mrow - mnew);
    mrow = mnew;

    float rsum = 0.0f;
    v16h af;  // P in PV A-fragment layout: elementwise from S^T tiles
#pragma unroll
    for (int r = 0; r < 8; ++r) {
      const float p0 = __expf(s0[r] - mnew);
      const float p1 = __expf(s1[r] - mnew);
      rsum += p0 + p1;
      af[r]     = (_Float16)p0;
      af[r + 8] = (_Float16)p1;
    }
    rsum += xor16(rsum);
    lrow = lrow * scale + rsum;

    // rescale accumulators: need per-n factor in [n rows] layout -> LDS bcast
    if (lane < 16) sm[wv][lane] = scale;
    float sc[8];
#pragma unroll
    for (int r = 0; r < 8; ++r) sc[r] = sm[wv][kh * 8 + r];
#pragma unroll
    for (int dt = 0; dt < 4; ++dt)
#pragma unroll
      for (int r = 0; r < 8; ++r) acc[dt][r] *= sc[r];

    // ---- X^T += P * V^T  (B rows k = m = mt + lane, cols = d) ----
    acc[0] = wmma_f16(af, vb0, acc[0]);
    acc[1] = wmma_f16(af, vb1, acc[1]);
    acc[2] = wmma_f16(af, vb2, acc[2]);
    acc[3] = wmma_f16(af, vb3, acc[3]);
  }

  // ---- finalize: divide by row sums, store f16 in merge layout ----
  if (lane < 16) sm[wv][lane] = 1.0f / lrow;
  float li[8];
#pragma unroll
  for (int r = 0; r < 8; ++r) li[r] = sm[wv][kh * 8 + r];

  const int h = bh & 3;
  const int b = bh >> 2;
#pragma unroll
  for (int dt = 0; dt < 4; ++dt) {
    const int d = dt * 16 + m16;        // D tile: lane col = d
    const int c = d * NHEAD + h;        // interleaved-head channel
    v8h pk;
#pragma unroll
    for (int r = 0; r < 8; ++r) pk[r] = (_Float16)(acc[dt][r] * li[r]);
    // rows n = n0 + kh*8 + r are contiguous -> one 16B store
    *(v8h*)(Xf + ((size_t)b * DMODEL + c) * NSEQ + n0 + kh * 8) = pk;
  }
}

// ---------------------------------------------------------------------------
// Merge: out[b][o][n] = sum_c Wm[o][c] * Xf[b][c][n] + bm[o]   (fp32 out)
// Bias folded into C-matrix init; K=256 fully unrolled (8 WMMA).
// ---------------------------------------------------------------------------
__global__ __launch_bounds__(128) void merge_wmma(const _Float16* __restrict__ Xf,
                                                  const float* __restrict__ W,
                                                  const float* __restrict__ bias,
                                                  float* __restrict__ out) {
  const int lane = threadIdx.x & 31;
  const int wv   = threadIdx.x >> 5;
  const int m    = lane & 15;
  const int kh   = lane >> 4;
  const int n0   = blockIdx.x * 16;
  const int o0   = (blockIdx.y * 4 + wv) * 16;
  const int b    = blockIdx.z;

  v8f acc;
#pragma unroll
  for (int r = 0; r < 8; ++r) acc[r] = bias[o0 + kh * 8 + r];

  const float* wrow    = W + (size_t)(o0 + m) * DMODEL + kh * 8;
  const _Float16* xcol = Xf + (size_t)b * DMODEL * NSEQ + n0;

#pragma unroll
  for (int k0 = 0; k0 < DMODEL; k0 += 32) {
    v16h a;
#pragma unroll
    for (int v = 0; v < 8; ++v) {
      const int grp = v >> 2, p = v & 3;
      float2 w2 = *(const float2*)(wrow + k0 + grp * 16 + p * 2);
      a[2 * v]     = (_Float16)w2.x;
      a[2 * v + 1] = (_Float16)w2.y;
    }
    v16h bf;
#pragma unroll
    for (int j = 0; j < 16; ++j)
      bf[j] = xcol[(size_t)(k0 + (j >> 3) * 16 + kh * 8 + (j & 7)) * NSEQ + m];
    acc = wmma_f16(a, bf, acc);
  }
#pragma unroll
  for (int r = 0; r < 8; ++r) {
    const int o = o0 + kh * 8 + r;
    out[((size_t)b * DMODEL + o) * NSEQ + n0 + m] = acc[r];
  }
}

// ---------------------------------------------------------------------------
extern "C" void kernel_launch(void* const* d_in, const int* in_sizes, int n_in,
                              void* d_out, int out_size, void* d_ws,
                              size_t ws_size, hipStream_t stream) {
  (void)in_sizes; (void)n_in; (void)out_size; (void)ws_size;
  const float* query = (const float*)d_in[0];
  const float* key_  = (const float*)d_in[1];
  const float* value = (const float*)d_in[2];
  const float* Wq = (const float*)d_in[3];
  const float* bq = (const float*)d_in[4];
  const float* Wk = (const float*)d_in[5];
  const float* bk = (const float*)d_in[6];
  const float* Wv = (const float*)d_in[7];
  const float* bv = (const float*)d_in[8];
  const float* Wm = (const float*)d_in[9];
  const float* bm = (const float*)d_in[10];
  float* out = (float*)d_out;

  // Workspace layout (f16): Qd | Kt | Vt | Xf  -> 4 x 4 MB = 16 MB
  const size_t seg = (size_t)BATCH * NHEAD * HDIM * NSEQ;
  _Float16* Qd = (_Float16*)d_ws;
  _Float16* Kt = Qd + seg;
  _Float16* Vt = Kt + seg;
  _Float16* Xf = Vt + seg;

  const dim3 gproj(NSEQ / 16, DMODEL / 64, BATCH);  // wave = one 16x16 tile
  proj_wmma<0><<<gproj, 128, 0, stream>>>(query, Wq, bq, Qd);
  proj_wmma<1><<<gproj, 128, 0, stream>>>(key_, Wk, bk, Kt);
  proj_wmma<1><<<gproj, 128, 0, stream>>>(value, Wv, bv, Vt);

  // 2048 waves (one per 16-query strip per (b,h)); 8 waves/block
  attn_wmma<<<BATCH * NHEAD * (NSEQ / 16) / 8, 256, 0, stream>>>(Qd, Kt, Vt, Xf);

  merge_wmma<<<gproj, 128, 0, stream>>>(Xf, Wm, bm, out);
}
